// SoftSplit_77738908058568
// MI455X (gfx1250) — matrix operand order repaired and checked by reading, not compiled, not run on hardware
//
#include <hip/hip_runtime.h>
#include <stdint.h>

typedef __attribute__((ext_vector_type(16))) _Float16 v16h;
typedef __attribute__((ext_vector_type(8)))  float    v8f;

union Frag16 { uint32_t u[8]; v16h h; };
union H2     { _Float16 h[2]; uint32_t u; };

#define C1   64
#define C2   96
#define HW   128
#define CIN  576   // 64*3*3

// One workgroup = one (b,h) image row: 128 pixels x 96 outputs.
// 8 waves of 32; wave wtile owns pixels [wtile*16, wtile*16+16).
__global__ __launch_bounds__(256)
void softsplit_wmma(const float* __restrict__ x,
                    const float* __restrict__ W,
                    const float* __restrict__ bias,
                    float* __restrict__ out)
{
    // x tile: 3 rows x 32 channel-pairs x 130 cols (zero-padded), packed f16x2
    __shared__ uint32_t ldsX[3 * 32 * 130];
    // W tap slice: 32 channel-pairs x 96 outputs, packed f16x2
    __shared__ uint32_t ldsW[32 * 96];

    const int tid  = threadIdx.x;
    const int bh   = blockIdx.x;        // b*128 + h
    const int bimg = bh >> 7;
    const int h    = bh & (HW - 1);

    // ---- stage x rows h-1..h+1 into LDS as packed f16 channel pairs ----
    const float* xb = x + (size_t)bimg * C1 * HW * HW;
    for (int i = tid; i < 3 * 32 * 130; i += 256) {
        int col = i % 130;              // padded column 0..129
        int t   = i / 130;
        int p   = t % 32;               // channel pair
        int row = t / 32;               // 0..2 -> hr = h-1+row
        int hr  = h - 1 + row;
        H2 v; v.u = 0u;
        if (hr >= 0 && hr < HW && col >= 1 && col <= HW) {
            int w = col - 1;
            v.h[0] = (_Float16)xb[((size_t)(2 * p    ) * HW + hr) * HW + w];
            v.h[1] = (_Float16)xb[((size_t)(2 * p + 1) * HW + hr) * HW + w];
        }
        ldsX[i] = v.u;
    }

    const int lane  = tid & 31;
    const int wtile = tid >> 5;         // 0..7 : which 16-pixel tile
    const int lhalf = lane >> 4;        // lane half (ISA fragment layouts)
    const int r     = lane & 15;

    // bias for this lane's output column in each of the 6 N-tiles
    float bval[6];
    #pragma unroll
    for (int nt = 0; nt < 6; ++nt) bval[nt] = bias[nt * 16 + r];

    v8f acc[6] = {};

    for (int pos = 0; pos < 9; ++pos) {          // kernel tap = (ki,kj)
        const int ki = pos / 3;
        const int kj = pos % 3;

        __syncthreads();                          // previous tap's B reads done
        // ---- stage W[:, c*9+pos] for c=0..63 as packed channel pairs ----
        for (int i = tid; i < 32 * 96; i += 256) {
            int n = i % 96;
            int p = i / 96;                       // channel pair
            H2 v;
            v.h[0] = (_Float16)W[(size_t)n * CIN + (2 * p    ) * 9 + pos];
            v.h[1] = (_Float16)W[(size_t)n * CIN + (2 * p + 1) * 9 + pos];
            ldsW[p * 96 + n] = v.u;
        }
        __syncthreads();

        #pragma unroll
        for (int kh = 0; kh < 2; ++kh) {          // channels 0-31 / 32-63
            // A fragment: 16x32 f16, ISA layout (K pairs per VGPR)
            Frag16 a;
            const int colA = wtile * 16 + r + kj; // padded col, 0..129
            #pragma unroll
            for (int v = 0; v < 8; ++v) {
                int kpb = (v < 4 ? 2 * v : 16 + 2 * (v - 4)) + lhalf * 8;
                int pairIdx = kh * 16 + (kpb >> 1);
                a.u[v] = ldsX[(ki * 32 + pairIdx) * 130 + colA];
            }
            #pragma unroll
            for (int nt = 0; nt < 6; ++nt) {
                // B fragment: 32x16 f16, one output column per lane
                Frag16 bf;
                #pragma unroll
                for (int v = 0; v < 8; ++v) {
                    int pairIdx = kh * 16 + (v < 4 ? v : 8 + (v - 4)) + lhalf * 4;
                    bf.u[v] = ldsW[pairIdx * 96 + nt * 16 + r];
                }
                acc[nt] = __builtin_amdgcn_wmma_f32_16x16x32_f16(
                    /*neg_a=*/false, a.h, /*neg_b=*/false, bf.h,
                    /*c_mod=*/(short)0, acc[nt],
                    /*reuse_a=*/false, /*reuse_b=*/false);
            }
        }
    }

    // ---- epilogue: bias + store. D layout: VGPR j -> row j+8*lhalf, col r ----
    const size_t outBase = (size_t)bh * HW * C2;
    #pragma unroll
    for (int nt = 0; nt < 6; ++nt) {
        #pragma unroll
        for (int j = 0; j < 8; ++j) {
            int w = wtile * 16 + j + 8 * lhalf;
            out[outBase + (size_t)w * C2 + nt * 16 + r] = acc[nt][j] + bval[nt];
        }
    }
}

extern "C" void kernel_launch(void* const* d_in, const int* in_sizes, int n_in,
                              void* d_out, int out_size, void* d_ws, size_t ws_size,
                              hipStream_t stream) {
    const float* x    = (const float*)d_in[0];   // (8, 64, 128, 128)
    const float* W    = (const float*)d_in[1];   // (96, 576)
    const float* bias = (const float*)d_in[2];   // (96,)
    float* out = (float*)d_out;                  // (8, 16384, 96)

    dim3 grid(8 * HW);                           // one block per (b, h)
    dim3 block(256);
    softsplit_wmma<<<grid, block, 0, stream>>>(x, W, bias, out);
}
